// Correlation_9122510536790
// MI455X (gfx1250) — matrix operand order repaired and checked by reading, not compiled, not run on hardware
//
#include <hip/hip_runtime.h>

// CDNA5 / gfx1250 correlation layer via v_wmma_f32_16x16x32_bf16 with
// bf16 hi/lo split-precision (3 WMMAs per K-step) for ~fp32 accuracy.
// Round 2: batched B-fragment ds_loads, 3 independent accumulator chains,
// branch-free diagonal scatter, vectorized float4 staging.

typedef __attribute__((ext_vector_type(16))) __bf16 v16bf;
typedef __attribute__((ext_vector_type(8)))  __bf16 v8bf;
typedef __attribute__((ext_vector_type(8)))  float  v8f;
typedef __attribute__((ext_vector_type(4)))  float  v4f;

#define Bn 16
#define Cc 128
#define Hh 96
#define Ww 160
#define ND 9          // 9 dx values per dy
#define XT 32         // x-positions per workgroup (2 M-tiles of 16)
#define PA 136        // padded K pitch (bf16 elems), multiple of 8 -> 16B rows
#define PB 136
#define NW 48         // staged in2 columns: x0-4 .. x0+43 (3 N-windows of 16)
#define NVEC 12       // NW / 4

union Frag { v16bf v; v8bf h[2]; };

#define WMMA_BF16(A, B, C) \
  __builtin_amdgcn_wmma_f32_16x16x32_bf16(false, (A), false, (B), (short)0, (C), false, false)

__global__ __launch_bounds__(128) void corr_wmma_kernel(
    const float* __restrict__ in1, const float* __restrict__ in2,
    float* __restrict__ out) {
  const int x0   = blockIdx.x * XT;
  const int y    = blockIdx.y;
  const int b    = blockIdx.z;
  const int tid  = threadIdx.x;
  const int lane = tid & 31;
  const int w    = tid >> 5;            // wave id 0..3

  __shared__ __attribute__((aligned(16))) __bf16 sAhi[XT * PA];
  __shared__ __attribute__((aligned(16))) __bf16 sAlo[XT * PA];
  __shared__ __attribute__((aligned(16))) __bf16 sBhi[NW * PB];
  __shared__ __attribute__((aligned(16))) __bf16 sBlo[NW * PB];
  __shared__ float sOut[ND * XT + 4];   // +dump slot for invalid diagonals

  // ---------------- stage in1 tile: [x][k] bf16 hi/lo (vec4 loads) --------
  const float* in1p = in1 + ((size_t)(b * Cc) * Hh + y) * Ww + x0;
  for (int i = tid; i < Cc * (XT / 4); i += 128) {
    int c = i >> 3;                  // channel
    int v = i & 7;                   // vec4 index within the 32-wide tile
    v4f f = *(const v4f*)(in1p + (size_t)c * (Hh * Ww) + v * 4);
    #pragma unroll
    for (int e = 0; e < 4; ++e) {
      __bf16 hi = (__bf16)f[e];
      __bf16 lo = (__bf16)(f[e] - (float)hi);
      sAhi[(v * 4 + e) * PA + c] = hi;
      sAlo[(v * 4 + e) * PA + c] = lo;
    }
  }
  __syncthreads();

  // ---------------- persistent A fragments (dy-invariant) ----------------
  // wave -> (M-tile, N-window): w0:(0,0) w1:(0,1) w2:(1,1) w3:(1,2)
  const int mt    = (w >= 2) ? 1 : 0;
  const int wi    = (w == 0) ? 0 : ((w == 3) ? 2 : 1);
  const int m16   = lane & 15;
  const int khalf = lane >> 4;

  Frag Ahi[4], Alo[4];
  {
    const int xl = mt * 16 + m16;                    // A row (M) per lane
    #pragma unroll
    for (int ks = 0; ks < 4; ++ks) {
      // 16-bit A layout: lane half selects k runs {8h..8h+7} and {16+8h..}
      int off = xl * PA + ks * 32 + khalf * 8;
      const v8bf* ph = (const v8bf*)&sAhi[off];
      const v8bf* pl = (const v8bf*)&sAlo[off];
      Ahi[ks].h[0] = ph[0];  Ahi[ks].h[1] = ph[2];   // +0 and +16 elems
      Alo[ks].h[0] = pl[0];  Alo[ks].h[1] = pl[2];
    }
  }

  const float* in2b = in2 + (size_t)b * Cc * Hh * Ww;
  const int ncol = wi * 16 + m16;                    // B column (N) per lane

  for (int dy = -4; dy <= 4; ++dy) {
    __syncthreads();                  // prior readers of sB*/sOut are done
    const int  y2    = y + dy;
    const int  y2c   = y2 < 0 ? 0 : (y2 >= Hh ? Hh - 1 : y2);
    const bool rowok = (y2 >= 0) && (y2 < Hh);

    // -------- stage in2 row y2: [x'][k] bf16 hi/lo, clamped vec4 loads ----
    // Window is 4-aligned and W%4==0, so any vec4 with a valid element is
    // fully in-bounds; clamping only keeps fully-masked pointers legal.
    for (int i = tid; i < Cc * NVEC; i += 128) {
      int c  = i / NVEC;
      int v  = i - c * NVEC;
      int xg = x0 - 4 + v * 4;
      int xc = xg < 0 ? 0 : (xg > Ww - 4 ? Ww - 4 : xg);
      v4f f = *(const v4f*)(in2b + ((size_t)c * Hh + y2c) * Ww + xc);
      #pragma unroll
      for (int e = 0; e < 4; ++e) {
        bool ok  = rowok & ((unsigned)(xg + e) < (unsigned)Ww);
        float fe = ok ? f[e] : 0.0f;
        __bf16 hi = (__bf16)fe;
        __bf16 lo = (__bf16)(fe - (float)hi);
        sBhi[(v * 4 + e) * PB + c] = hi;
        sBlo[(v * 4 + e) * PB + c] = lo;
      }
    }
    __syncthreads();

    // -------- preload ALL B fragments, then a clean 12-WMMA block ---------
    Frag Bh[4], Bl[4];
    #pragma unroll
    for (int ks = 0; ks < 4; ++ks) {
      // 16-bit B layout: lane half selects k runs {8h..8h+7} and {16+8h..}
      int off = ncol * PB + ks * 32 + khalf * 8;
      const v8bf* ph = (const v8bf*)&sBhi[off];
      const v8bf* pl = (const v8bf*)&sBlo[off];
      Bh[ks].h[0] = ph[0];  Bh[ks].h[1] = ph[2];
      Bl[ks].h[0] = pl[0];  Bl[ks].h[1] = pl[2];
    }

    // 3 independent accumulator chains (split terms) -> dep distance 3
    v8f a0 = {}, a1 = {}, a2 = {};
    #pragma unroll
    for (int ks = 0; ks < 4; ++ks) {
      a0 = WMMA_BF16(Ahi[ks].v, Bh[ks].v, a0);   // main product
      a1 = WMMA_BF16(Alo[ks].v, Bh[ks].v, a1);   // A low-order correction
      a2 = WMMA_BF16(Ahi[ks].v, Bl[ks].v, a2);   // B low-order correction
    }

    // -------- branch-free diagonal scatter into sOut ----------------------
    #pragma unroll
    for (int v = 0; v < 8; ++v) {
      int M  = v + 8 * khalf;                    // C/D layout row
      int mx = mt * 16 + M;                      // x within 32-block
      int dx = -4 + wi * 16 + m16 - mx;          // x' - x
      float r = (a0[v] + a1[v] + a2[v]) * (1.0f / 128.0f);
      int idx = ((unsigned)(dx + 4) <= 8u) ? (dx + 4) * XT + mx : ND * XT;
      sOut[idx] = r;                             // invalid lanes -> dump slot
    }
    __syncthreads();

    // -------- coalesced store: 9 rows of 32 contiguous floats -------------
    const int dbase = (dy + 4) * ND;
    for (int i = tid; i < ND * XT; i += 128) {
      int dxi = i >> 5;
      int xl  = i & 31;
      out[(((size_t)b * (ND * ND) + dbase + dxi) * Hh + y) * Ww + x0 + xl] = sOut[i];
    }
  }
}

extern "C" void kernel_launch(void* const* d_in, const int* in_sizes, int n_in,
                              void* d_out, int out_size, void* d_ws, size_t ws_size,
                              hipStream_t stream) {
  (void)in_sizes; (void)n_in; (void)d_ws; (void)ws_size; (void)out_size;
  const float* in1 = (const float*)d_in[0];
  const float* in2 = (const float*)d_in[1];
  float* out = (float*)d_out;
  dim3 grid(Ww / XT, Hh, Bn);   // 5 x 96 x 16
  dim3 block(128);
  corr_wmma_kernel<<<grid, block, 0, stream>>>(in1, in2, out);
}
